// Spatial_Temporal_Heterogeneous_Graph_68899865362918
// MI455X (gfx1250) — matrix operand order repaired and checked by reading, not compiled, not run on hardware
//
#include <hip/hip_runtime.h>
#include <hip/hip_bf16.h>
#include <math.h>

#define BDIM 128
#define TDIM 32
#define NPERB 64
#define NNODES (BDIM * NPERB)   // 8192

typedef __attribute__((ext_vector_type(16))) _Float16 v16h;
typedef __attribute__((ext_vector_type(8)))  _Float16 v8h;
typedef __attribute__((ext_vector_type(8)))  float    v8f;

// ---------------- WMMA helpers (gfx1250 wave32, 16x16x32 f16 -> f32) ----------------

__device__ __forceinline__ v8f wmma32(v16h a, v16h b, v8f c) {
  return __builtin_amdgcn_wmma_f32_16x16x32_f16(false, a, false, b, (short)0, c,
                                                false, false);
}

// A fragment (16x32 f16) from X[rows][ld] (row-major, LDS or global f16).
// lane<16: row=lane,   elems 0..15 = K kb..kb+15 (kb=k0)
// lane>=16: row=lane-16, kb=k0+16.  Both halves 16B contiguous -> 2x b128 loads.
__device__ __forceinline__ v16h ld_A_h(const _Float16* X, int ld, int m0, int k0) {
  int lane = threadIdx.x & 31;
  int r = m0 + (lane & 15);
  int kb = k0 + ((lane & 16) ? 16 : 0);
  v8h lo = *(const v8h*)(X + r * ld + kb);
  v8h hi = *(const v8h*)(X + r * ld + kb + 8);
  v16h a;
#pragma unroll
  for (int i = 0; i < 8; ++i) { a[i] = lo[i]; a[8 + i] = hi[i]; }
  return a;
}

// B fragment (32x16) = W^T tile from f16 [n][k] array (ld = ldw), 2x b128 loads.
__device__ __forceinline__ v16h ld_B_h(const _Float16* Wh, int ldw, int n0, int k0) {
  int lane = threadIdx.x & 31;
  int n = n0 + (lane & 15);
  int kb = k0 + ((lane & 16) ? 16 : 0);
  const v8h* p = (const v8h*)(Wh + n * ldw + kb);
  v8h lo = p[0], hi = p[1];
  v16h b;
#pragma unroll
  for (int i = 0; i < 8; ++i) { b[i] = lo[i]; b[8 + i] = hi[i]; }
  return b;
}

// B fragment straight from global f32 W[n][k] (full-size, no padding needed).
__device__ __forceinline__ v16h ld_B_f32(const float* W, int ldw, int n0, int k0) {
  int lane = threadIdx.x & 31;
  int n = n0 + (lane & 15);
  int kb = k0 + ((lane & 16) ? 16 : 0);
  const float4* p = (const float4*)(W + n * ldw + kb);
  float4 c0 = p[0], c1 = p[1], c2 = p[2], c3 = p[3];
  v16h b;
  b[0] = (_Float16)c0.x;  b[1] = (_Float16)c0.y;  b[2] = (_Float16)c0.z;  b[3] = (_Float16)c0.w;
  b[4] = (_Float16)c1.x;  b[5] = (_Float16)c1.y;  b[6] = (_Float16)c1.z;  b[7] = (_Float16)c1.w;
  b[8] = (_Float16)c2.x;  b[9] = (_Float16)c2.y;  b[10] = (_Float16)c2.z; b[11] = (_Float16)c2.w;
  b[12] = (_Float16)c3.x; b[13] = (_Float16)c3.y; b[14] = (_Float16)c3.z; b[15] = (_Float16)c3.w;
  return b;
}

// Store a D tile column-major: Os[n][m] with leading dim ldm. Each lane owns one
// column (fixed n), 8 contiguous m values -> two float4 (b128) stores.
__device__ __forceinline__ void st_D_cm(float* Os, int ldm, int n, int mb, v8f acc,
                                        float bv) {
  float* p = Os + n * ldm + mb;
  *(float4*)(p)     = make_float4(acc[0] + bv, acc[1] + bv, acc[2] + bv, acc[3] + bv);
  *(float4*)(p + 4) = make_float4(acc[4] + bv, acc[5] + bv, acc[6] + bv, acc[7] + bv);
}

// Stage global f32 W[nrows][kdim] into LDS f16 Ws[64][64], zero-padded, branchless.
__device__ __forceinline__ void stage_w(_Float16* Ws, const float* W, int nrows, int kdim) {
  for (int idx = threadIdx.x; idx < 64 * 64; idx += 128) {
    int n = idx >> 6, k = idx & 63;
    int nc = n < nrows ? n : nrows - 1;
    int kc = k < kdim ? k : kdim - 1;
    float w = W[nc * kdim + kc];
    Ws[idx] = (_Float16)((n < nrows && k < kdim) ? w : 0.0f);
  }
}

// 64-row GEMM in one 4-wave workgroup. Output COLUMN-MAJOR: Os[n][m], ld 64.
__device__ __forceinline__ void gemm_tile64(const _Float16* Xs, const _Float16* Ws,
                                            const float* bias, int kdim, int dout,
                                            float* Os) {
  int wave = threadIdx.x >> 5;
  int lane = threadIdx.x & 31;
  int m0 = wave * 16;
  int ksteps = (kdim + 31) >> 5;
  int nts = (dout + 15) >> 4;
#pragma unroll
  for (int nt = 0; nt < nts; ++nt) {
    v8f acc = {};
#pragma unroll
    for (int kt = 0; kt < ksteps; ++kt) {
      acc = wmma32(ld_A_h(Xs, 64, m0, kt * 32), ld_B_h(Ws, 64, nt * 16, kt * 32), acc);
    }
    int n = nt * 16 + (lane & 15);
    int mb = m0 + ((lane & 16) ? 8 : 0);
    if (n < dout) st_D_cm(Os, 64, n, mb, acc, bias[n]);
  }
}

// Same but stores f16 output, column-major (one b128 store per tile per lane).
__device__ __forceinline__ void gemm_tile64_h(const _Float16* Xs, const _Float16* Ws,
                                              const float* bias, _Float16* Os) {
  int wave = threadIdx.x >> 5;
  int lane = threadIdx.x & 31;
  int m0 = wave * 16;
#pragma unroll
  for (int nt = 0; nt < 4; ++nt) {
    v8f acc = {};
#pragma unroll
    for (int kt = 0; kt < 2; ++kt) {
      acc = wmma32(ld_A_h(Xs, 64, m0, kt * 32), ld_B_h(Ws, 64, nt * 16, kt * 32), acc);
    }
    int n = nt * 16 + (lane & 15);
    int mb = m0 + ((lane & 16) ? 8 : 0);
    float bv = bias[n];
    v8h hv;
#pragma unroll
    for (int i = 0; i < 8; ++i) hv[i] = (_Float16)(acc[i] + bv);
    *(v8h*)(Os + n * 64 + mb) = hv;
  }
}

// One SubGraph layer (Hs is COLUMN-MAJOR [chan][row]).
__device__ __forceinline__ void mlp_layer(_Float16* Xs, float* Hs, _Float16* Ws, float* red,
                                          const float* w1, const float* b1,
                                          const float* g, const float* be,
                                          const float* w2, const float* b2,
                                          int din, int dout) {
  const int tid = threadIdx.x;
  stage_w(Ws, w1, 64, din);
  __syncthreads();
  gemm_tile64(Xs, Ws, b1, din, 64, Hs);
  __syncthreads();
  stage_w(Ws, w2, dout, 64);          // overlap w2 staging with LayerNorm
  if (tid < 64) {                     // per-row LayerNorm + ReLU -> Xs (row-major f16)
    float mu = 0.f;
    for (int c = 0; c < 64; ++c) mu += Hs[c * 64 + tid];
    mu *= (1.0f / 64.0f);
    float var = 0.f;
    for (int c = 0; c < 64; ++c) { float d = Hs[c * 64 + tid] - mu; var += d * d; }
    var *= (1.0f / 64.0f);
    float inv = rsqrtf(var + 1e-5f);
    for (int c = 0; c < 64; ++c) {
      float h = (Hs[c * 64 + tid] - mu) * inv * g[c] + be[c];
      Xs[tid * 64 + c] = (_Float16)fmaxf(h, 0.f);
    }
  }
  __syncthreads();
  gemm_tile64(Xs, Ws, b2, 64, dout, Hs);
  __syncthreads();
  if (tid < dout) {                   // segment mean: contiguous per-thread read
    float s = 0.f;
    for (int r = 0; r < 64; ++r) s += Hs[tid * 64 + r];
    red[tid] = s * (1.0f / 64.0f);
  }
  __syncthreads();
  if (tid < 64) {                     // concat([mean, own]), pad K to 32
    for (int c = 0; c < dout; ++c) Xs[tid * 64 + c] = (_Float16)red[c];
    for (int c = 0; c < dout; ++c) Xs[tid * 64 + dout + c] = (_Float16)Hs[c * 64 + tid];
    for (int c = 2 * dout; c < 32; ++c) Xs[tid * 64 + c] = (_Float16)0.0f;
  }
  __syncthreads();
}

// ---------------- Kernel 0: zero-fill the dense A output ----------------
__global__ void zero_fill4(float4* __restrict__ p, size_t n4) {
  size_t i = (size_t)blockIdx.x * blockDim.x + threadIdx.x;
  if (i < n4) p[i] = make_float4(0.f, 0.f, 0.f, 0.f);
}

// ---------------- Kernel 1: fused SubGraph + masked attention ----------------
__global__ __launch_bounds__(128) void stg_fused_sub_attn(
    const float* __restrict__ x_neighbor, const float* __restrict__ x_bbox,
    const float* m0_w1, const float* m0_b1, const float* m0_g, const float* m0_be,
    const float* m0_w2, const float* m0_b2,
    const float* m1_w1, const float* m1_b1, const float* m1_g, const float* m1_be,
    const float* m1_w2, const float* m1_b2,
    const float* m2_w1, const float* m2_b1, const float* m2_g, const float* m2_be,
    const float* m2_w2, const float* m2_b2,
    const float* bm_w1, const float* bm_b1, const float* bm_g, const float* bm_be,
    const float* bm_w2, const float* bm_b2,
    const float* q_w, const float* q_b, const float* k_w, const float* k_b,
    const float* v_w, const float* v_b,
    _Float16* __restrict__ gg, float* __restrict__ Aout) {
  __shared__ _Float16 Xs[64 * 64];   // activations, row-major
  __shared__ float    Hs[64 * 64];   // GEMM outputs, column-major [chan][row]
  __shared__ _Float16 Ws[64 * 64];   // staged weights
  __shared__ _Float16 Ks[64 * 64];   // k, column-major [chan][row]
  __shared__ float    Vs[64 * 64];   // v, column-major [chan][row]
  __shared__ float    brow[64];
  __shared__ float    tmp64[64];
  __shared__ float    qrow[64];
  __shared__ float    red[64];
  __shared__ float    sred[2];

  const int tid = threadIdx.x;
  const int b = blockIdx.x / TDIM;
  const int t = blockIdx.x % TDIM;

  // Load x_neighbor rows for this (b,t): [64 nodes][5], padded to 32 K-cols.
  if (tid < 64) {
    const float* xr = x_neighbor + ((size_t)(b * NPERB + tid) * TDIM + t) * 5;
#pragma unroll
    for (int i = 0; i < 5; ++i) Xs[tid * 64 + i] = (_Float16)xr[i];
    for (int i = 5; i < 32; ++i) Xs[tid * 64 + i] = (_Float16)0.0f;
  }
  __syncthreads();

  mlp_layer(Xs, Hs, Ws, red, m0_w1, m0_b1, m0_g, m0_be, m0_w2, m0_b2, 5, 8);
  mlp_layer(Xs, Hs, Ws, red, m1_w1, m1_b1, m1_g, m1_be, m1_w2, m1_b2, 16, 16);
  mlp_layer(Xs, Hs, Ws, red, m2_w1, m2_b1, m2_g, m2_be, m2_w2, m2_b2, 32, 32);
  // Xs now holds x [64][64] (row-major f16).

  // k = x @ k_w^T + k_b -> Ks (f16, column-major)
  stage_w(Ws, k_w, 64, 64);
  __syncthreads();
  gemm_tile64_h(Xs, Ws, k_b, Ks);
  __syncthreads();
  // v = x @ v_w^T + v_b -> Vs (f32, column-major)
  stage_w(Ws, v_w, 64, 64);
  __syncthreads();
  gemm_tile64(Xs, Ws, v_b, 64, 64, Vs);

  // bbox path: Linear(4->64) -> LN -> ReLU -> Linear(64->64), then q projection
  if (tid < 64) {
    const float* xb = x_bbox + (size_t)(b * TDIM + t) * 4;
    float s = bm_b1[tid];
#pragma unroll
    for (int i = 0; i < 4; ++i) s += xb[i] * bm_w1[tid * 4 + i];
    brow[tid] = s;
  }
  __syncthreads();
  if (tid == 0) {
    float mu = 0.f;
    for (int c = 0; c < 64; ++c) mu += brow[c];
    mu *= (1.0f / 64.0f);
    float var = 0.f;
    for (int c = 0; c < 64; ++c) { float d = brow[c] - mu; var += d * d; }
    var *= (1.0f / 64.0f);
    sred[0] = mu;
    sred[1] = rsqrtf(var + 1e-5f);
  }
  __syncthreads();
  if (tid < 64)
    tmp64[tid] = fmaxf((brow[tid] - sred[0]) * sred[1] * bm_g[tid] + bm_be[tid], 0.f);
  __syncthreads();
  if (tid < 64) {
    float s = bm_b2[tid];
    for (int j = 0; j < 64; ++j) s += tmp64[j] * bm_w2[tid * 64 + j];
    brow[tid] = s;  // bbox_emb
  }
  __syncthreads();
  if (tid < 64) {
    float s = q_b[tid];
    for (int j = 0; j < 64; ++j) s += brow[j] * q_w[tid * 64 + j];
    qrow[tid] = s;  // q
  }
  __syncthreads();

  // masked exp-attention over this block's 64 neighbors (Ks column-major)
  if (tid < 64) {
    float s = 0.f;
    for (int c = 0; c < 64; ++c) s += qrow[c] * (float)Ks[c * 64 + tid];
    red[tid] = expf(s * 0.125f);  // / sqrt(64)
  }
  __syncthreads();
  if (tid == 0) {
    float s = 0.f;
    for (int r = 0; r < 64; ++r) s += red[r];
    sred[0] = s;
  }
  __syncthreads();
  if (tid < 64) {
    float a = red[tid] / sred[0];
    Aout[((size_t)(t * BDIM + b)) * NNODES + (size_t)b * NPERB + tid] = a;
    tmp64[tid] = a;
  }
  __syncthreads();
  if (tid < 64) {  // gg[b,t,c] = sum_r A[r] * v[r,c]  (Vs column-major: contiguous)
    float s = 0.f;
    for (int r = 0; r < 64; ++r) s += tmp64[r] * Vs[tid * 64 + r];
    gg[(size_t)(b * TDIM + t) * 64 + tid] = (_Float16)s;
  }
}

// ---------------- Kernel 2a: ego embedding relu(x_ego @ lw^T + lb) -> f16 ----------
__global__ void ego_embed(const float* __restrict__ x_ego, const float* __restrict__ lw,
                          const float* __restrict__ lb, _Float16* __restrict__ e) {
  int i = blockIdx.x * blockDim.x + threadIdx.x;
  if (i >= BDIM * TDIM * 64) return;
  int row = i >> 6, c = i & 63;
  const float* xr = x_ego + (size_t)row * 4;
  float s = lb[c];
#pragma unroll
  for (int j = 0; j < 4; ++j) s += xr[j] * lw[c * 4 + j];
  e[i] = (_Float16)fmaxf(s, 0.f);
}

// ---------------- Kernel 2b: gi = X[M,64](f16) @ W[192,64]^T + bias ------------------
__global__ __launch_bounds__(128) void gemm_gates(const _Float16* __restrict__ X,
                                                  const float* __restrict__ W,
                                                  const float* __restrict__ bias,
                                                  float* __restrict__ OUT, int M) {
  int tile = blockIdx.x * 4 + (threadIdx.x >> 5);  // one 16x16 tile per wave
  const int NT = 12;                               // 192 / 16
  int mt = tile / NT, nt = tile % NT;
  if (mt * 16 >= M) return;  // wave-uniform
  int lane = threadIdx.x & 31;
  v8f acc = {};
#pragma unroll
  for (int kt = 0; kt < 2; ++kt) {
    acc = wmma32(ld_A_h(X, 64, mt * 16, kt * 32), ld_B_f32(W, 64, nt * 16, kt * 32), acc);
  }
  int n = nt * 16 + (lane & 15);
  int mb = mt * 16 + ((lane & 16) ? 8 : 0);
  float bv = bias[n];
#pragma unroll
  for (int i = 0; i < 8; ++i) OUT[(size_t)(mb + i) * 192 + n] = acc[i] + bv;
}

// ---------------- Kernel 3: sequential GRUs (neigh + ego), 32 batch rows / WG --------
__global__ __launch_bounds__(128) void gru_seq(
    const float* __restrict__ gi_n, const float* __restrict__ gi_e,
    const float* __restrict__ wh_n, const float* __restrict__ bh_n,
    const float* __restrict__ wh_e, const float* __restrict__ bh_e,
    const int* __restrict__ tptr, float* __restrict__ out) {
  __shared__ _Float16 Hh[32 * 64];   // h (f16, row-major) for WMMA A-frags
  __shared__ float    Hf[32 * 64];   // h (f32)
  __shared__ float    GH[192 * 32];  // gh staging, column-major [gate-chan][row]
  __shared__ _Float16 Whs[192 * 64]; // wh staged once (f16)
  const int tid = threadIdx.x;
  const int which = blockIdx.x >> 2;     // 0 = neigh GRU, 1 = ego GRU
  const int b0 = (blockIdx.x & 3) * 32;  // batch-row slice
  const float* gi = which ? gi_e : gi_n;
  const float* wh = which ? wh_e : wh_n;
  const float* bh = which ? bh_e : bh_n;
  const int outoff = which ? 64 : 0;
  int tmax = *tptr;
  if (tmax >= TDIM) tmax = TDIM - 1;
  if (tmax < 0) tmax = 0;

  for (int i = tid; i < 192 * 64; i += 128) Whs[i] = (_Float16)wh[i];
  for (int i = tid; i < 32 * 64; i += 128) { Hf[i] = 0.f; Hh[i] = (_Float16)0.f; }
  __syncthreads();
  const int wave = tid >> 5, lane = tid & 31;

  for (int step = 0; step <= tmax; ++step) {
    // gh = h @ wh^T + bh : [32,192], 24 tiles, 6 per wave, K=64 (2 wmma each)
#pragma unroll
    for (int i = 0; i < 6; ++i) {
      int tile = wave * 6 + i;
      int mt = tile / 12, nt = tile % 12;
      v8f acc = {};
#pragma unroll
      for (int kt = 0; kt < 2; ++kt) {
        acc = wmma32(ld_A_h(Hh, 64, mt * 16, kt * 32), ld_B_h(Whs, 64, nt * 16, kt * 32), acc);
      }
      int n = nt * 16 + (lane & 15);
      int mb = mt * 16 + ((lane & 16) ? 8 : 0);
      st_D_cm(GH, 32, n, mb, acc, bh[n]);  // column-major, 2x b128
    }
    __syncthreads();
    // gates (PyTorch order r, z, n); GH is [gate-chan][row]
    for (int e = tid; e < 32 * 64; e += 128) {
      int lr = e >> 6, c = e & 63;
      const float* gir = gi + (size_t)((b0 + lr) * TDIM + step) * 192;
      float r = 1.f / (1.f + expf(-(gir[c] + GH[c * 32 + lr])));
      float z = 1.f / (1.f + expf(-(gir[64 + c] + GH[(64 + c) * 32 + lr])));
      float nn = tanhf(gir[128 + c] + r * GH[(128 + c) * 32 + lr]);
      float h = (1.f - z) * nn + z * Hf[e];
      Hf[e] = h;
      Hh[e] = (_Float16)h;
    }
    __syncthreads();
  }
  for (int e = tid; e < 32 * 64; e += 128) {
    int lr = e >> 6, c = e & 63;
    out[(size_t)(b0 + lr) * 128 + outoff + c] = Hf[e];
  }
}

// ---------------- launch ----------------
extern "C" void kernel_launch(void* const* d_in, const int* in_sizes, int n_in,
                              void* d_out, int out_size, void* d_ws, size_t ws_size,
                              hipStream_t stream) {
  (void)in_sizes; (void)n_in; (void)out_size; (void)ws_size;
  const float* x_neighbor = (const float*)d_in[0];
  const float* x_bbox     = (const float*)d_in[1];
  const float* x_ego      = (const float*)d_in[2];
  const int*   tptr       = (const int*)d_in[3];
  const float* m0p[6]; const float* m1p[6]; const float* m2p[6]; const float* bmp[6];
  for (int i = 0; i < 6; ++i) m0p[i] = (const float*)d_in[4 + i];
  for (int i = 0; i < 6; ++i) m1p[i] = (const float*)d_in[10 + i];
  for (int i = 0; i < 6; ++i) m2p[i] = (const float*)d_in[16 + i];
  for (int i = 0; i < 6; ++i) bmp[i] = (const float*)d_in[22 + i];
  const float* q_w = (const float*)d_in[28]; const float* q_b = (const float*)d_in[29];
  const float* k_w = (const float*)d_in[30]; const float* k_b = (const float*)d_in[31];
  const float* v_w = (const float*)d_in[32]; const float* v_b = (const float*)d_in[33];
  const float* gru_wi = (const float*)d_in[34]; const float* gru_wh = (const float*)d_in[35];
  const float* gru_bi = (const float*)d_in[36]; const float* gru_bh = (const float*)d_in[37];
  const float* ego_lw = (const float*)d_in[38]; const float* ego_lb = (const float*)d_in[39];
  const float* ego_gru_wi = (const float*)d_in[40]; const float* ego_gru_wh = (const float*)d_in[41];
  const float* ego_gru_bi = (const float*)d_in[42]; const float* ego_gru_bh = (const float*)d_in[43];

  float* out  = (float*)d_out;
  float* Aout = out + (size_t)BDIM * 128;  // A region: [T,B,N] f32

  // workspace layout (bytes): gg_h 512KB | ee_h 512KB | gi_n 3MB | gi_e 3MB
  char* wsb = (char*)d_ws;
  _Float16* gg_h = (_Float16*)(wsb);
  _Float16* ee_h = (_Float16*)(wsb + (512 << 10));
  float*    gi_n = (float*)(wsb + (1024 << 10));
  float*    gi_e = (float*)(wsb + (4096 << 10));

  // K0: zero-fill A (nonzero blocks overwritten by K1)
  size_t n4 = (size_t)TDIM * BDIM * NNODES / 4;
  zero_fill4<<<(unsigned)((n4 + 255) / 256), 256, 0, stream>>>((float4*)Aout, n4);

  // K1: fused subgraph + attention
  stg_fused_sub_attn<<<BDIM * TDIM, 128, 0, stream>>>(
      x_neighbor, x_bbox,
      m0p[0], m0p[1], m0p[2], m0p[3], m0p[4], m0p[5],
      m1p[0], m1p[1], m1p[2], m1p[3], m1p[4], m1p[5],
      m2p[0], m2p[1], m2p[2], m2p[3], m2p[4], m2p[5],
      bmp[0], bmp[1], bmp[2], bmp[3], bmp[4], bmp[5],
      q_w, q_b, k_w, k_b, v_w, v_b, gg_h, Aout);

  // K2a: ego embedding (f16)
  ego_embed<<<(BDIM * TDIM * 64 + 255) / 256, 256, 0, stream>>>(x_ego, ego_lw, ego_lb, ee_h);

  // K2b: input-gate GEMMs (3072 tiles, 4 waves/block)
  gemm_gates<<<768, 128, 0, stream>>>(gg_h, gru_wi, gru_bi, gi_n, BDIM * TDIM);
  gemm_gates<<<768, 128, 0, stream>>>(ee_h, ego_gru_wi, ego_gru_bi, gi_e, BDIM * TDIM);

  // K3: sequential GRUs
  gru_seq<<<8, 128, 0, stream>>>(gi_n, gi_e, gru_wh, gru_bh, ego_gru_wh, ego_gru_bh,
                                 tptr, out);
}
